// CausalSelfAttention_55370718380602
// MI455X (gfx1250) — matrix operand order repaired and checked by reading, not compiled, not run on hardware
//
#include <hip/hip_runtime.h>
#include <hip/hip_bf16.h>

// ---------------------------------------------------------------------------
// CDNA5 (gfx1250) causal self-attention, bf16 WMMA pipeline, software-pipelined.
// All matmuls use v_wmma_f32_16x16x32_bf16 (fp32 accumulate).
// ---------------------------------------------------------------------------

typedef __attribute__((ext_vector_type(16))) __bf16 v16bf;
typedef __attribute__((ext_vector_type(8)))  float  v8f;
typedef __attribute__((ext_vector_type(4)))  unsigned int u32x4;

#define BATCH     2
#define SEQ       2048
#define DIM       1024
#define NHEADS    16
#define HEADDIM   64
#define MTOT      (BATCH*SEQ)      // 4096
#define N_QKV     (3*DIM)          // 3072

union Frag { v16bf v; u32x4 q[2]; };

__device__ __forceinline__ unsigned short f32_to_bf16(float f) {
    unsigned u = __float_as_uint(f);
    u += 0x7FFFu + ((u >> 16) & 1u);           // round-to-nearest-even
    return (unsigned short)(u >> 16);
}

// A fragment: 16x32 bf16, row-major source, ld = row stride (elements).
// ISA 7.12.2: lane half h, row m=lane&15 holds K = h*8+0..7 and 16+h*8+0..7.
__device__ __forceinline__ v16bf load_a16x32(const unsigned short* src, int ld) {
    const int lane = threadIdx.x & 31;
    const int m = lane & 15, h = lane >> 4;
    const unsigned short* r = src + m * ld + h * 8;
    Frag f;
    f.q[0] = *(const u32x4*)(r);
    f.q[1] = *(const u32x4*)(r + 16);
    return f.v;
}

// B fragment: 32x16 (KxN) from an "n-major" source Bt[n][k] (ld = row stride).
// Lane holds column n=lane&15, K = h*8+0..7 and 16+h*8+0..7 (matches A).
__device__ __forceinline__ v16bf load_bt32x16(const unsigned short* src, int ld) {
    const int lane = threadIdx.x & 31;
    const int n = lane & 15, h = lane >> 4;
    const unsigned short* r = src + n * ld + h * 8;
    Frag f;
    f.q[0] = *(const u32x4*)(r);
    f.q[1] = *(const u32x4*)(r + 16);
    return f.v;
}

__device__ __forceinline__ v8f wmma_bf16(v16bf a, v16bf b, v8f c) {
    return __builtin_amdgcn_wmma_f32_16x16x32_bf16(
        false, a, false, b, (short)0, c, false, false);
}

#define V8F_ZERO (v8f){0.f,0.f,0.f,0.f,0.f,0.f,0.f,0.f}

// ---------------------------------------------------------------------------
// Converters
// ---------------------------------------------------------------------------
__global__ void convert_f32_bf16(const float* __restrict__ in,
                                 unsigned short* __restrict__ out, int n) {
    int i = blockIdx.x * blockDim.x + threadIdx.x;
    if (i < n) out[i] = f32_to_bf16(in[i]);
}

// out[c*R + r] = bf16(in[r*C + c])  (transpose so GEMM B operands are [N,K])
__global__ void transpose_f32_bf16(const float* __restrict__ in,
                                   unsigned short* __restrict__ out, int R, int C) {
    int i = blockIdx.x * blockDim.x + threadIdx.x;
    if (i < R * C) {
        int r = i / C, c = i - r * C;
        out[c * R + r] = f32_to_bf16(in[i]);
    }
}

// ---------------------------------------------------------------------------
// Kernel 1: QKV projection + RoPE epilogue.  Software-pipelined K loop
// (unroll 2 so the double buffers rotate by register renaming, no moves).
// Block = 128 threads (4 waves); block tile 128(M)x64(N); wave tile 32x64.
// Q,K -> [BH,S,hd] bf16 (RoPE applied).  V -> transposed [BH,hd,S] bf16.
// ---------------------------------------------------------------------------
__global__ __launch_bounds__(128) void gemm_qkv_rope(
    const unsigned short* __restrict__ xb,
    const unsigned short* __restrict__ wqkvT,
    unsigned short* __restrict__ qws,
    unsigned short* __restrict__ kws,
    unsigned short* __restrict__ vtws) {
    const int w    = threadIdx.x >> 5;
    const int lane = threadIdx.x & 31;
    const int n15  = lane & 15, h = lane >> 4;
    const int n0   = blockIdx.x * 64;
    const int m0   = blockIdx.y * 128 + w * 32;

    v8f acc[2][4];
#pragma unroll
    for (int mi = 0; mi < 2; ++mi)
#pragma unroll
        for (int j = 0; j < 4; ++j) acc[mi][j] = V8F_ZERO;

    v16bf aC[2], bC[4];
#pragma unroll
    for (int mi = 0; mi < 2; ++mi)
        aC[mi] = load_a16x32(xb + (m0 + mi * 16) * DIM, DIM);
#pragma unroll
    for (int j = 0; j < 4; ++j)
        bC[j] = load_bt32x16(wqkvT + (n0 + j * 16) * DIM, DIM);

#pragma unroll 2
    for (int k0 = 0; k0 < DIM; k0 += 32) {
        const int kn = (k0 + 32 < DIM) ? k0 + 32 : k0;   // dummy refetch at end
        v16bf aN[2], bN[4];
#pragma unroll
        for (int mi = 0; mi < 2; ++mi)
            aN[mi] = load_a16x32(xb + (m0 + mi * 16) * DIM + kn, DIM);
#pragma unroll
        for (int j = 0; j < 4; ++j)
            bN[j] = load_bt32x16(wqkvT + (n0 + j * 16) * DIM + kn, DIM);
#pragma unroll
        for (int mi = 0; mi < 2; ++mi)
#pragma unroll
            for (int j = 0; j < 4; ++j)
                acc[mi][j] = wmma_bf16(aC[mi], bC[j], acc[mi][j]);
#pragma unroll
        for (int mi = 0; mi < 2; ++mi) aC[mi] = aN[mi];
#pragma unroll
        for (int j = 0; j < 4; ++j) bC[j] = bN[j];
    }

    const int section = n0 / DIM;              // 0=Q 1=K 2=V
    const int head    = (n0 % DIM) / HEADDIM;  // tile covers exactly one head

#pragma unroll
    for (int mi = 0; mi < 2; ++mi)
#pragma unroll
    for (int i = 0; i < 8; ++i) {
        const int m  = m0 + mi * 16 + i + 8 * h;
        const int b  = m >> 11;
        const int s  = m & (SEQ - 1);
        const int bh = (b << 4) + head;
        if (section == 2) {
#pragma unroll
            for (int j = 0; j < 4; ++j) {
                const int c = j * 16 + n15;
                vtws[((unsigned)(bh * HEADDIM + c)) * SEQ + s] =
                    f32_to_bf16(acc[mi][j][i]);
            }
        } else {
            unsigned short* dst = (section == 0) ? qws : kws;
#pragma unroll
            for (int j = 0; j < 4; ++j) {
                const int c   = j * 16 + n15;
                float val  = acc[mi][j][i];
                float part = acc[mi][j ^ 2][i];             // rotate-half partner
                float rot  = (c < 32) ? -part : part;
                const int fidx = c & 31;
                float inv = __expf(-0.2878231366f * (float)fidx); // 10000^(-f/32)
                float ang = (float)s * inv, sv, cv;
                __sincosf(ang, &sv, &cv);
                dst[((unsigned)(bh * SEQ + s)) * HEADDIM + c] =
                    f32_to_bf16(val * cv + rot * sv);
            }
        }
    }
}

// ---------------------------------------------------------------------------
// Kernel 2: flash attention (causal, online softmax), one (b,h,qtile) / block.
// Block = 128 threads (4 waves); each wave owns 16 query rows.
// V loads + next-K prefetch issued BEFORE the softmax VALU block so global
// latency hides under exp/shuffle work; LDS fence only guards the P transpose.
// ---------------------------------------------------------------------------
__global__ __launch_bounds__(128) void flash_attn(
    const unsigned short* __restrict__ qws,
    const unsigned short* __restrict__ kws,
    const unsigned short* __restrict__ vtws,
    unsigned short* __restrict__ aws) {
    __shared__ __align__(16) unsigned short plds[4][16 * 32];   // per-wave P tile

    const int w    = threadIdx.x >> 5;
    const int lane = threadIdx.x & 31;
    const int n15  = lane & 15, h = lane >> 4;
    const int bh   = blockIdx.x >> 5;
    const int qt   = blockIdx.x & 31;
    const int m0   = qt * 64 + w * 16;      // wave-uniform

    const unsigned short* qb = qws  + ((unsigned)(bh * SEQ + m0)) * HEADDIM;
    const unsigned short* kb = kws  + ((unsigned)bh) * SEQ * HEADDIM;
    const unsigned short* vb = vtws + ((unsigned)bh) * HEADDIM * SEQ;

    const v16bf aq0 = load_a16x32(qb, HEADDIM);
    const v16bf aq1 = load_a16x32(qb + 32, HEADDIM);

    v8f o[4];
#pragma unroll
    for (int j = 0; j < 4; ++j) o[j] = V8F_ZERO;
    float rm[8], rs[8];
#pragma unroll
    for (int i = 0; i < 8; ++i) { rm[i] = -3.0e38f; rs[i] = 0.f; }
    const float scale = 0.125f;   // 1/sqrt(64)

    // preload K fragments for kt = 0
    v16bf kf[4];
    kf[0] = load_bt32x16(kb, HEADDIM);
    kf[1] = load_bt32x16(kb + 32, HEADDIM);
    kf[2] = load_bt32x16(kb + 16 * HEADDIM, HEADDIM);
    kf[3] = load_bt32x16(kb + 16 * HEADDIM + 32, HEADDIM);

    for (int kt = 0; kt * 32 <= m0 + 15; ++kt) {
        const int key0 = kt * 32;
        // ---- scores S = Q * K^T (16 x 32) --------------------------------
        v8f s0 = V8F_ZERO, s1 = V8F_ZERO;
        s0 = wmma_bf16(aq0, kf[0], s0);
        s0 = wmma_bf16(aq1, kf[1], s0);
        s1 = wmma_bf16(aq0, kf[2], s1);
        s1 = wmma_bf16(aq1, kf[3], s1);

        // ---- issue V loads for this tile + K prefetch for next tile ------
        v16bf vf[4];
#pragma unroll
        for (int jn = 0; jn < 4; ++jn)
            vf[jn] = load_bt32x16(vb + (unsigned)(jn * 16) * SEQ + key0, SEQ);
        const int keyN = ((kt + 1) * 32 <= m0 + 15) ? (kt + 1) * 32 : key0;
        kf[0] = load_bt32x16(kb + (unsigned)keyN * HEADDIM, HEADDIM);
        kf[1] = load_bt32x16(kb + (unsigned)keyN * HEADDIM + 32, HEADDIM);
        kf[2] = load_bt32x16(kb + (unsigned)(keyN + 16) * HEADDIM, HEADDIM);
        kf[3] = load_bt32x16(kb + (unsigned)(keyN + 16) * HEADDIM + 32, HEADDIM);

        // ---- mask + online softmax (rows: m = m0+i+8h, cols: lane&15) ----
#pragma unroll
        for (int i = 0; i < 8; ++i) {
            const int m  = m0 + i + 8 * h;
            const int nA = key0 + n15, nB = key0 + 16 + n15;
            float v0 = (nA <= m) ? s0[i] * scale : -3.0e38f;
            float v1 = (nB <= m) ? s1[i] * scale : -3.0e38f;
            float mx = fmaxf(v0, v1);
            mx = fmaxf(mx, __shfl_xor(mx, 8));
            mx = fmaxf(mx, __shfl_xor(mx, 4));
            mx = fmaxf(mx, __shfl_xor(mx, 2));
            mx = fmaxf(mx, __shfl_xor(mx, 1));     // max over 16-lane half
            float nm    = fmaxf(rm[i], mx);
            float alpha = __expf(rm[i] - nm);
            float p0 = __expf(v0 - nm);
            float p1 = __expf(v1 - nm);
            float ps = p0 + p1;
            ps += __shfl_xor(ps, 8);
            ps += __shfl_xor(ps, 4);
            ps += __shfl_xor(ps, 2);
            ps += __shfl_xor(ps, 1);
            rs[i] = rs[i] * alpha + ps;
            rm[i] = nm;
            o[0][i] *= alpha; o[1][i] *= alpha; o[2][i] *= alpha; o[3][i] *= alpha;
            const int r = i + 8 * h;               // C-layout row -> LDS row
            plds[w][r * 32 + n15]      = f32_to_bf16(p0);
            plds[w][r * 32 + 16 + n15] = f32_to_bf16(p1);
        }
        // C-layout -> A-layout via per-wave LDS tile (DS in-order per wave;
        // wait + compiler barrier guard the round-trip).
        asm volatile("s_wait_dscnt 0" ::: "memory");
        v16bf pa = load_a16x32(&plds[w][0], 32);
        asm volatile("" ::: "memory");

        // ---- O += P * V --------------------------------------------------
#pragma unroll
        for (int jn = 0; jn < 4; ++jn)
            o[jn] = wmma_bf16(pa, vf[jn], o[jn]);
    }

    const int head = bh & 15, b = bh >> 4;
#pragma unroll
    for (int i = 0; i < 8; ++i) {
        const float inv = 1.0f / rs[i];
        const int s = m0 + i + 8 * h;
        const unsigned base = ((unsigned)(b * SEQ + s)) * DIM + head * HEADDIM;
#pragma unroll
        for (int jn = 0; jn < 4; ++jn)
            aws[base + jn * 16 + n15] = f32_to_bf16(o[jn][i] * inv);
    }
}

// ---------------------------------------------------------------------------
// Kernel 3: output projection -> fp32 d_out.  Same pipelined 32x64 wave tile.
// ---------------------------------------------------------------------------
__global__ __launch_bounds__(128) void gemm_out(
    const unsigned short* __restrict__ aws,
    const unsigned short* __restrict__ woutT,
    float* __restrict__ out) {
    const int w    = threadIdx.x >> 5;
    const int lane = threadIdx.x & 31;
    const int n15  = lane & 15, h = lane >> 4;
    const int n0   = blockIdx.x * 64;
    const int m0   = blockIdx.y * 128 + w * 32;

    v8f acc[2][4];
#pragma unroll
    for (int mi = 0; mi < 2; ++mi)
#pragma unroll
        for (int j = 0; j < 4; ++j) acc[mi][j] = V8F_ZERO;

    v16bf aC[2], bC[4];
#pragma unroll
    for (int mi = 0; mi < 2; ++mi)
        aC[mi] = load_a16x32(aws + (m0 + mi * 16) * DIM, DIM);
#pragma unroll
    for (int j = 0; j < 4; ++j)
        bC[j] = load_bt32x16(woutT + (n0 + j * 16) * DIM, DIM);

#pragma unroll 2
    for (int k0 = 0; k0 < DIM; k0 += 32) {
        const int kn = (k0 + 32 < DIM) ? k0 + 32 : k0;
        v16bf aN[2], bN[4];
#pragma unroll
        for (int mi = 0; mi < 2; ++mi)
            aN[mi] = load_a16x32(aws + (m0 + mi * 16) * DIM + kn, DIM);
#pragma unroll
        for (int j = 0; j < 4; ++j)
            bN[j] = load_bt32x16(woutT + (n0 + j * 16) * DIM + kn, DIM);
#pragma unroll
        for (int mi = 0; mi < 2; ++mi)
#pragma unroll
            for (int j = 0; j < 4; ++j)
                acc[mi][j] = wmma_bf16(aC[mi], bC[j], acc[mi][j]);
#pragma unroll
        for (int mi = 0; mi < 2; ++mi) aC[mi] = aN[mi];
#pragma unroll
        for (int j = 0; j < 4; ++j) bC[j] = bN[j];
    }
#pragma unroll
    for (int mi = 0; mi < 2; ++mi)
#pragma unroll
    for (int i = 0; i < 8; ++i) {
        const int m = m0 + mi * 16 + i + 8 * h;
#pragma unroll
        for (int j = 0; j < 4; ++j)
            out[(unsigned)m * DIM + n0 + j * 16 + n15] = acc[mi][j][i];
    }
}

// ---------------------------------------------------------------------------
// Host launcher
// ---------------------------------------------------------------------------
extern "C" void kernel_launch(void* const* d_in, const int* in_sizes, int n_in,
                              void* d_out, int out_size, void* d_ws, size_t ws_size,
                              hipStream_t stream) {
    (void)in_sizes; (void)n_in; (void)out_size; (void)ws_size;
    const float* x    = (const float*)d_in[0];
    const float* wqkv = (const float*)d_in[1];
    const float* wout = (const float*)d_in[2];
    // d_in[3] = causal mask: implied by kernel, unused.

    unsigned short* xb    = (unsigned short*)d_ws;
    unsigned short* wqkvT = xb    + (size_t)MTOT * DIM;
    unsigned short* woutT = wqkvT + (size_t)N_QKV * DIM;
    unsigned short* qws   = woutT + (size_t)DIM * DIM;
    unsigned short* kws   = qws   + (size_t)BATCH * NHEADS * SEQ * HEADDIM;
    unsigned short* vtws  = kws   + (size_t)BATCH * NHEADS * SEQ * HEADDIM;
    unsigned short* aws   = vtws  + (size_t)BATCH * NHEADS * SEQ * HEADDIM;
    // total workspace: 48 MiB of bf16

    {
        int n = MTOT * DIM;
        convert_f32_bf16<<<(n + 255) / 256, 256, 0, stream>>>(x, xb, n);
    }
    {
        int n = DIM * N_QKV;
        transpose_f32_bf16<<<(n + 255) / 256, 256, 0, stream>>>(wqkv, wqkvT, DIM, N_QKV);
    }
    {
        int n = DIM * DIM;
        transpose_f32_bf16<<<(n + 255) / 256, 256, 0, stream>>>(wout, woutT, DIM, DIM);
    }
    gemm_qkv_rope<<<dim3(N_QKV / 64, MTOT / 128), 128, 0, stream>>>(
        xb, wqkvT, qws, kws, vtws);
    flash_attn<<<BATCH * NHEADS * (SEQ / 64), 128, 0, stream>>>(
        qws, kws, vtws, aws);
    gemm_out<<<dim3(DIM / 64, MTOT / 128), 128, 0, stream>>>(
        aws, woutT, (float*)d_out);
}
